// color_invariant_quadruplet_19361712570611
// MI455X (gfx1250) — matrix operand order repaired and checked by reading, not compiled, not run on hardware
//
#include <hip/hip_runtime.h>
#include <hip/hip_bf16.h>

typedef __attribute__((ext_vector_type(16))) _Float16 v16h;
typedef __attribute__((ext_vector_type(8)))  float    v8f;

// ---------------------------------------------------------------------------
// Phase 1: per h-edge label byte: zlab[j] = z[src_g[src_h[j]]] | (z[dst_g[dst_h[j]]]<<3)
// ---------------------------------------------------------------------------
__global__ void build_labels_kernel(const int* __restrict__ z,
                                    const int* __restrict__ src_g,
                                    const int* __restrict__ dst_g,
                                    const int* __restrict__ src_h,
                                    const int* __restrict__ dst_h,
                                    unsigned char* __restrict__ zlab,
                                    int EH) {
    int j = blockIdx.x * blockDim.x + threadIdx.x;
    if (j < EH) {
        int zs = z[src_g[src_h[j]]];
        int zd = z[dst_g[dst_h[j]]];
        zlab[j] = (unsigned char)(zs | (zd << 3));
    }
}

// ---------------------------------------------------------------------------
// Phase 2: build T[64][64] = S(64x12) @ E(12x64) via WMMA in LDS, then
// stream output rows: 16 lanes per edge, one float4 per lane (coalesced 256B).
// ---------------------------------------------------------------------------
__global__ __launch_bounds__(256) void quad_embed_kernel(
        const int* __restrict__ src_i, const int* __restrict__ dst_i,
        const unsigned char* __restrict__ zlab,
        const float* __restrict__ e1, const float* __restrict__ e2,
        const float* __restrict__ e3, const float* __restrict__ e4,
        const float* __restrict__ e5, const float* __restrict__ e6,
        float* __restrict__ out, int EI) {

    __shared__ __align__(16) float Tbl[64 * 64];   // 16 KB LUT

    const float* et[6] = { e1, e2, e3, e4, e5, e6 };

    const int lane = threadIdx.x & 31;
    const int wave = threadIdx.x >> 5;   // 0..7 (8 waves per block)
    const int g    = lane >> 4;          // lane half (0/1)
    const int mn   = lane & 15;          // row/col within tile

    // ---- Table build: 16 tiles (4x4 of 16x16), 2 tiles per wave ----
    #pragma unroll
    for (int ti = 0; ti < 2; ++ti) {
        const int t  = wave * 2 + ti;    // 0..15
        const int mt = t >> 2;           // code-tile
        const int nt = t & 3;            // feature-tile
        const int code = mt * 16 + mn;   // A-matrix row for this lane
        const int f    = nt * 16 + mn;   // B/C column for this lane

        // A: 16x32 f16 selector. half h -> K = h + 16*g (matches B)
        v16h a;
        #pragma unroll
        for (int h = 0; h < 16; ++h) {
            const int v = h >> 1, p = h & 1;
            const int K = h + 16 * g;
            float aval = 0.0f;
            if (K < 12) {
                // selector column K pairs with table K>>1, row K&1
                aval = (((code >> (K >> 1)) & 1) == (K & 1)) ? 1.0f : 0.0f;
            }
            a[h] = (_Float16)aval;
        }

        // B: 32x16 f16, half h -> K = h + 16*g ; hi/lo split of fp32 E rows
        v16h bhi, blo;
        #pragma unroll
        for (int h = 0; h < 16; ++h) {
            float val = 0.0f;
            if (g == 0 && h < 12) {
                val = et[h >> 1][(h & 1) * 64 + f];   // E row h, column f
            }
            const _Float16 hi = (_Float16)val;
            bhi[h] = hi;
            blo[h] = (_Float16)(val - (float)hi);
        }

        v8f c = {};
        c = __builtin_amdgcn_wmma_f32_16x16x32_f16(false, a, false, bhi,
                                                   (short)0, c, false, false);
        c = __builtin_amdgcn_wmma_f32_16x16x32_f16(false, a, false, blo,
                                                   (short)0, c, false, false);

        // C/D layout: VGPR v holds row v (lanes 0-15) / v+8 (lanes 16-31)
        #pragma unroll
        for (int v = 0; v < 8; ++v) {
            Tbl[(mt * 16 + v + 8 * g) * 64 + f] = c[v];
        }
    }
    __syncthreads();

    // ---- Streaming phase: 16 threads per edge, float4 per thread ----
    const long long total  = (long long)EI * 16;           // float4 slots
    const long long stride = (long long)gridDim.x * blockDim.x;
    float4* __restrict__ out4 = (float4*)out;

    for (long long idx = (long long)blockIdx.x * blockDim.x + threadIdx.x;
         idx < total; idx += stride) {
        const int e   = (int)(idx >> 4);
        const int sub = (int)(idx & 15);

        const int la = zlab[src_i[e]];
        const int lb = zlab[dst_i[e]];
        const int za = la & 7, zc = la >> 3;
        const int zb = lb & 7, zd = lb >> 3;

        const int code = ((za == zc) ? 1  : 0) |
                         ((za == zb) ? 2  : 0) |
                         ((zc == zb) ? 4  : 0) |
                         ((za == zd) ? 8  : 0) |
                         ((zc == zd) ? 16 : 0) |
                         ((zb == zd) ? 32 : 0);

        out4[idx] = ((const float4*)(Tbl + code * 64))[sub];
    }
}

// ---------------------------------------------------------------------------
extern "C" void kernel_launch(void* const* d_in, const int* in_sizes, int n_in,
                              void* d_out, int out_size, void* d_ws, size_t ws_size,
                              hipStream_t stream) {
    const int* z     = (const int*)d_in[0];
    const int* src_g = (const int*)d_in[1];
    const int* dst_g = (const int*)d_in[2];
    const int* src_h = (const int*)d_in[3];
    const int* dst_h = (const int*)d_in[4];
    const int* src_i = (const int*)d_in[5];
    const int* dst_i = (const int*)d_in[6];
    const float* e1  = (const float*)d_in[7];
    const float* e2  = (const float*)d_in[8];
    const float* e3  = (const float*)d_in[9];
    const float* e4  = (const float*)d_in[10];
    const float* e5  = (const float*)d_in[11];
    const float* e6  = (const float*)d_in[12];

    const int E_H = in_sizes[3];
    const int E_I = in_sizes[5];

    unsigned char* zlab = (unsigned char*)d_ws;   // E_H bytes (~1.5 MB)

    build_labels_kernel<<<(E_H + 255) / 256, 256, 0, stream>>>(
        z, src_g, dst_g, src_h, dst_h, zlab, E_H);

    float* out = (float*)d_out;
    const int blocks = 4096;   // grid-stride; amortizes per-block LUT build
    quad_embed_kernel<<<blocks, 256, 0, stream>>>(
        src_i, dst_i, zlab, e1, e2, e3, e4, e5, e6, out, E_I);
}
